// DTD_75943611727948
// MI455X (gfx1250) — matrix operand order repaired and checked by reading, not compiled, not run on hardware
//
#include <hip/hip_runtime.h>
#include <hip/hip_bf16.h>

// ---------------- problem constants ----------------
#define NB     128
#define NC     256
#define NHW    1024   // 8*128 spatial
#define NT     64
#define NCLASS 128
#define NHID   128    // per-direction LSTM hidden

// ---------------- WMMA types (CDNA5, wave32) ----------------
typedef __attribute__((ext_vector_type(16))) __bf16 v16bf;
typedef __attribute__((ext_vector_type(16))) short  v16s;
typedef __attribute__((ext_vector_type(8)))  float  v8f;

__device__ inline v8f vzero8() {
    v8f v = {0.f,0.f,0.f,0.f,0.f,0.f,0.f,0.f};
    return v;
}

// native hardware bf16 convert (v_cvt_pk_bf16_f32)
__device__ inline unsigned short f2bf(float f) {
    __bf16 h = (__bf16)f;
    return __builtin_bit_cast(unsigned short, h);
}
__device__ inline float bf2f(unsigned short s) {
    unsigned int u = ((unsigned int)s) << 16;
    return __builtin_bit_cast(float, u);
}
__device__ inline float sigm(float x) { return 1.0f / (1.0f + __expf(-x)); }

__device__ inline v8f wmma_bf(v16bf a, v16bf b, v8f c) {
    return __builtin_amdgcn_wmma_f32_16x16x32_bf16(
        false, a, false, b, (short)0, c, false, false);
}
__device__ inline v8f wmma_bf16s(v16s a, v16s b, v8f c) {
    return wmma_bf(__builtin_bit_cast(v16bf, a), __builtin_bit_cast(v16bf, b), c);
}

// ---- CDNA5 async global->LDS copy (ASYNCcnt-tracked), ISA §15.18.3 op 98 ----
__device__ inline void async_copy_b128(unsigned lds_byte_off, const void* gaddr) {
    asm volatile("global_load_async_to_lds_b128 %0, %1, off"
                 :: "v"(lds_byte_off), "v"(gaddr)
                 : "memory");
}
__device__ inline void wait_async0() {
    asm volatile("s_wait_asynccnt 0x0" ::: "memory");
}

// A-matrix 16x32 bf16 fragment per ISA layout.
// Arow = &A[m_row * K] with m_row = mt*16 + (lane&15). 2 contiguous 8-elem runs.
__device__ inline v16s load_a_bf16(const unsigned short* Arow, int k0, int lane) {
    int hi = (lane >= 16) ? 8 : 0;
    const unsigned short* p0 = Arow + k0 + hi;        // K = k0+hi .. +7
    const unsigned short* p1 = Arow + k0 + 16 + hi;   // K = k0+16+hi .. +7
    v16s r;
#pragma unroll
    for (int e = 0; e < 8; ++e) { r[e] = (short)p0[e]; r[8 + e] = (short)p1[e]; }
    return r;
}
// Same layout, f32 source with hardware bf16 convert.
__device__ inline v16bf load_a_f32(const float* Arow, int k0, int lane) {
    int hi = (lane >= 16) ? 8 : 0;
    const float* p0 = Arow + k0 + hi;
    const float* p1 = Arow + k0 + 16 + hi;
    v16bf r;
#pragma unroll
    for (int e = 0; e < 8; ++e) { r[e] = (__bf16)p0[e]; r[8 + e] = (__bf16)p1[e]; }
    return r;
}
// B-matrix 32x16 bf16 fragment. Brow = &B[n_col * K] (B stored [N][K], i.e. A*B^T).
__device__ inline v16s load_b_bf16(const unsigned short* Brow, int k0, int lane) {
    int hi = (lane >= 16) ? 8 : 0;
    const unsigned short* p0 = Brow + k0 + hi;
    const unsigned short* p1 = Brow + k0 + 16 + hi;
    v16s r;
#pragma unroll
    for (int e = 0; e < 8; ++e) { r[e] = (short)p0[e]; r[8 + e] = (short)p1[e]; }
    return r;
}
__device__ inline v16bf load_b_f32(const float* Brow, int k0, int lane) {
    int hi = (lane >= 16) ? 8 : 0;
    const float* p0 = Brow + k0 + hi;
    const float* p1 = Brow + k0 + 16 + hi;
    v16bf r;
#pragma unroll
    for (int e = 0; e < 8; ++e) { r[e] = (__bf16)p0[e]; r[8 + e] = (__bf16)p1[e]; }
    return r;
}

// ---------------- small prep kernels ----------------
__global__ void k_convert(const float* __restrict__ src, unsigned short* __restrict__ dst, int n) {
    int i = blockIdx.x * blockDim.x + threadIdx.x;
    if (i < n) dst[i] = f2bf(src[i]);
}

__global__ void k_lstm_bias(const float* __restrict__ bihf, const float* __restrict__ bhhf,
                            const float* __restrict__ bihb, const float* __restrict__ bhhb,
                            float* __restrict__ out) {
    int i = blockIdx.x * blockDim.x + threadIdx.x;
    if (i < 512) out[i] = bihf[i] + bhhf[i];
    else if (i < 1024) out[i] = bihb[i - 512] + bhhb[i - 512];
}

// prev_emb -> bf16 upper half of xs: xs[t][b][256+k]
__global__ void k_prev(const float* __restrict__ emb, const int* __restrict__ text,
                       unsigned short* __restrict__ xs) {
    int i = blockIdx.x * blockDim.x + threadIdx.x;  // over NT*NB*NC
    if (i >= NT * NB * NC) return;
    int k = i % NC;
    int b = (i / NC) % NB;
    int t = i / (NC * NB);
    int cls = (t == 0) ? 0 : text[b * NT + (t - 1)];
    xs[((size_t)t * NB + b) * (2 * NC) + NC + k] = f2bf(emb[cls * NC + k]);
}

// inverse spatial sums of A: one wave per (b,t)
__global__ void k_invsum(const float* __restrict__ A, float* __restrict__ invs) {
    int wid  = (blockIdx.x * blockDim.x + threadIdx.x) >> 5;  // b*NT + t
    int lane = threadIdx.x & 31;
    const float* p = A + (size_t)wid * NHW;
    float s = 0.f;
    for (int i = lane; i < NHW; i += 32) s += p[i];
#pragma unroll
    for (int off = 16; off; off >>= 1) s += __shfl_down(s, off, 32);
    if (lane == 0) invs[wid] = 1.0f / s;
}

// ---------------- attention pooling: C[t][b][c] = invsum * feature[b,c,:]·A[b,t,:] ----------------
// one wave per 16x64 strip (all 4 t-tiles). All fragments of a k-step are loaded/converted
// before the 4 WMMAs issue back-to-back (avoids WMMA->VALU WAR hazard NOPs).
__global__ void k_pool(const float* __restrict__ feat, const float* __restrict__ attn,
                       const float* __restrict__ invs, unsigned short* __restrict__ Cbf) {
    int wid  = (blockIdx.x * blockDim.x + threadIdx.x) >> 5;
    int lane = threadIdx.x & 31;
    int mt = wid & 15;  // 256/16 c-tiles
    int bb = wid >> 4;  // batch
    const float* arow = feat + ((size_t)bb * NC + (mt * 16 + (lane & 15))) * NHW;
    const float* brow[4];
    int tcol[4];
#pragma unroll
    for (int j = 0; j < 4; ++j) {
        tcol[j] = j * 16 + (lane & 15);
        brow[j] = attn + ((size_t)bb * NT + tcol[j]) * NHW;
    }
    v8f acc[4] = {vzero8(), vzero8(), vzero8(), vzero8()};
    for (int k0 = 0; k0 < NHW; k0 += 32) {
        v16bf a = load_a_f32(arow, k0, lane);
        v16bf bf[4];
#pragma unroll
        for (int j = 0; j < 4; ++j) bf[j] = load_b_f32(brow[j], k0, lane);
#pragma unroll
        for (int j = 0; j < 4; ++j) acc[j] = wmma_bf(a, bf[j], acc[j]);
    }
    int l_hi = lane >> 4;
#pragma unroll
    for (int j = 0; j < 4; ++j) {
        float sc = invs[bb * NT + tcol[j]];
#pragma unroll
        for (int r = 0; r < 8; ++r) {
            int c = mt * 16 + r + 8 * l_hi;
            Cbf[((size_t)tcol[j] * NB + bb) * NC + c] = f2bf(acc[j][r] * sc);
        }
    }
}

// ---------------- generic GEMM: C = A(bf16,[M][K]) * B(bf16,[N][K])^T + bias ----------------
// one wave per 16x64 output strip; N must be a multiple of 64.
__global__ void k_gemm(const unsigned short* __restrict__ A, const unsigned short* __restrict__ B,
                       const float* __restrict__ bias, float* __restrict__ Cf,
                       int M, int N, int K) {
    int wid  = (blockIdx.x * blockDim.x + threadIdx.x) >> 5;
    int lane = threadIdx.x & 31;
    int ntiles4 = N >> 6;
    int nt4 = wid % ntiles4;
    int mt  = wid / ntiles4;
    if (mt >= (M >> 4)) return;  // wave-uniform guard
    const unsigned short* Arow = A + (size_t)(mt * 16 + (lane & 15)) * K;
    int ncol[4];
    const unsigned short* Brow[4];
#pragma unroll
    for (int j = 0; j < 4; ++j) {
        ncol[j] = nt4 * 64 + j * 16 + (lane & 15);
        Brow[j] = B + (size_t)ncol[j] * K;
    }
    v8f acc[4];
#pragma unroll
    for (int j = 0; j < 4; ++j) {
        float bv = bias ? bias[ncol[j]] : 0.f;
#pragma unroll
        for (int r = 0; r < 8; ++r) acc[j][r] = bv;
    }
    for (int k0 = 0; k0 < K; k0 += 32) {
        v16s a = load_a_bf16(Arow, k0, lane);
        v16s bf[4];
#pragma unroll
        for (int j = 0; j < 4; ++j) bf[j] = load_b_bf16(Brow[j], k0, lane);
#pragma unroll
        for (int j = 0; j < 4; ++j) acc[j] = wmma_bf16s(a, bf[j], acc[j]);
    }
    int l_hi = lane >> 4;
#pragma unroll
    for (int j = 0; j < 4; ++j)
#pragma unroll
        for (int r = 0; r < 8; ++r) {
            int m = mt * 16 + r + 8 * l_hi;
            Cf[(size_t)m * N + ncol[j]] = acc[j][r];
        }
}

// ---------------- persistent bidirectional LSTM recurrence ----------------
// grid = 2 blocks (dir), 1024 threads. h (bf16) + Whh (bf16, async-staged) in LDS,
// c state in accumulator registers. Xproj already includes bih+bhh.
#define SMEM_H_BYTES (NB * NHID * 2)        // 32 KB
#define SMEM_W_BYTES (512 * NHID * 2)       // 128 KB
__global__ void __launch_bounds__(1024) k_lstm(const float* __restrict__ Xproj,
                                               const unsigned short* __restrict__ WhhBf,
                                               unsigned short* __restrict__ xs) {
    __shared__ __align__(16) char smem[SMEM_H_BYTES + SMEM_W_BYTES];  // 160 KB
    unsigned short* hsh = (unsigned short*)smem;
    unsigned short* wsh = (unsigned short*)(smem + SMEM_H_BYTES);
    int dir = blockIdx.x;
    const float* Xp = Xproj + (size_t)dir * NT * NB * 512;
    const char* wsrc = (const char*)(WhhBf + (size_t)dir * 512 * NHID);
    int tid = threadIdx.x, wid = tid >> 5, lane = tid & 31;

    // stage Whh into LDS with CDNA5 async copies (128 KB, 16 B per op)
    for (int i = tid * 16; i < SMEM_W_BYTES; i += 1024 * 16)
        async_copy_b128(SMEM_H_BYTES + i, wsrc + i);
    // zero h state
    for (int i = tid; i < NB * NHID; i += 1024) hsh[i] = 0;
    wait_async0();
    __syncthreads();

    int jt   = wid & 7;            // hidden j-tile (128/16)
    int mt0  = (wid >> 3) << 1;    // batch m-tiles {0,2,4,6}, 2 per wave
    int ncol = jt * 16 + (lane & 15);
    int l_hi = lane >> 4;
    v8f creg[2] = {vzero8(), vzero8()};  // c state lives in registers

    for (int step = 0; step < NT; ++step) {
        int t = dir ? (NT - 1 - step) : step;
        const float* xpt = Xp + (size_t)t * NB * 512;
        v8f acc[2][4];
#pragma unroll
        for (int mi = 0; mi < 2; ++mi)
#pragma unroll
            for (int g = 0; g < 4; ++g)
#pragma unroll
                for (int r = 0; r < 8; ++r) {
                    int b = (mt0 + mi) * 16 + r + 8 * l_hi;
                    acc[mi][g][r] = xpt[(size_t)b * 512 + g * NHID + ncol];
                }
#pragma unroll
        for (int k0 = 0; k0 < NHID; k0 += 32) {
            v16s afr0 = load_a_bf16(hsh + (size_t)(mt0 * 16 + (lane & 15)) * NHID, k0, lane);
            v16s afr1 = load_a_bf16(hsh + (size_t)((mt0 + 1) * 16 + (lane & 15)) * NHID, k0, lane);
            v16s bfr[4];
#pragma unroll
            for (int g = 0; g < 4; ++g)
                bfr[g] = load_b_bf16(wsh + (size_t)(g * NHID + ncol) * NHID, k0, lane);
#pragma unroll
            for (int g = 0; g < 4; ++g) {
                acc[0][g] = wmma_bf16s(afr0, bfr[g], acc[0][g]);
                acc[1][g] = wmma_bf16s(afr1, bfr[g], acc[1][g]);
            }
        }
        __syncthreads();  // all waves done reading h
#pragma unroll
        for (int mi = 0; mi < 2; ++mi) {
#pragma unroll
            for (int r = 0; r < 8; ++r) {
                int b = (mt0 + mi) * 16 + r + 8 * l_hi;
                float iv = sigm(acc[mi][0][r]);
                float fv = sigm(acc[mi][1][r]);
                float gv = tanhf(acc[mi][2][r]);
                float ov = sigm(acc[mi][3][r]);
                float cn = fv * creg[mi][r] + iv * gv;
                creg[mi][r] = cn;
                unsigned short hb = f2bf(ov * tanhf(cn));
                hsh[b * NHID + ncol] = hb;
                xs[((size_t)t * NB + b) * (2 * NC) + dir * NHID + ncol] = hb;
            }
        }
        __syncthreads();
    }
}

// ---------------- persistent GRU recurrence ----------------
// 1 block, 1024 threads. h in LDS bf16 [128][256] (64 KB). Whh_g (384 KB) stays in L2.
__global__ void __launch_bounds__(1024) k_gru(const float* __restrict__ gx,
                                              const unsigned short* __restrict__ WhhG,
                                              const float* __restrict__ bhh,
                                              unsigned short* __restrict__ hsOut) {
    __shared__ unsigned short hsh[NB * NC];  // 64 KB
    int tid = threadIdx.x, wid = tid >> 5, lane = tid & 31;
    for (int i = tid; i < NB * NC; i += 1024) hsh[i] = 0;
    __syncthreads();

    int jt   = wid & 15;           // 256/16 j-tiles
    int mt0  = (wid >> 4) << 2;    // batch tile group {0,4}, 4 per wave
    int ncol = jt * 16 + (lane & 15);
    int l_hi = lane >> 4;

    for (int t = 0; t < NT; ++t) {
        const float* gxt = gx + (size_t)t * NB * 768;
        v8f acc[4][3];
#pragma unroll
        for (int mi = 0; mi < 4; ++mi)
#pragma unroll
            for (int g = 0; g < 3; ++g) {
                float bv = bhh[g * NC + ncol];
#pragma unroll
                for (int r = 0; r < 8; ++r) acc[mi][g][r] = bv;
            }
#pragma unroll
        for (int k0 = 0; k0 < NC; k0 += 32) {
            v16s afr[4];
            v16s bfr[3];
#pragma unroll
            for (int mi = 0; mi < 4; ++mi)
                afr[mi] = load_a_bf16(hsh + (size_t)((mt0 + mi) * 16 + (lane & 15)) * NC, k0, lane);
#pragma unroll
            for (int g = 0; g < 3; ++g)
                bfr[g] = load_b_bf16(WhhG + (size_t)(g * NC + ncol) * NC, k0, lane);
#pragma unroll
            for (int g = 0; g < 3; ++g)
#pragma unroll
                for (int mi = 0; mi < 4; ++mi)
                    acc[mi][g] = wmma_bf16s(afr[mi], bfr[g], acc[mi][g]);
        }
        __syncthreads();
#pragma unroll
        for (int mi = 0; mi < 4; ++mi) {
#pragma unroll
            for (int r = 0; r < 8; ++r) {
                int b = (mt0 + mi) * 16 + r + 8 * l_hi;
                const float* gxb = gxt + (size_t)b * 768;
                float rv = sigm(gxb[ncol] + acc[mi][0][r]);
                float zv = sigm(gxb[NC + ncol] + acc[mi][1][r]);
                float nv = tanhf(gxb[2 * NC + ncol] + rv * acc[mi][2][r]);
                float hp = bf2f(hsh[b * NC + ncol]);
                float hn = (1.f - zv) * nv + zv * hp;
                unsigned short hb = f2bf(hn);
                hsh[b * NC + ncol] = hb;
                hsOut[((size_t)t * NB + b) * NC + ncol] = hb;
            }
        }
        __syncthreads();
    }
}

// ---------------- ragged output gather ----------------
__global__ void k_gather(const float* __restrict__ logits, const float* __restrict__ A,
                         const float* __restrict__ invs, const int* __restrict__ pb,
                         const int* __restrict__ pt, float* __restrict__ out, int L) {
    int i = blockIdx.x;
    int b = pb[i], t = pt[i];
    int tid = threadIdx.x;
    if (tid < NCLASS)
        out[(size_t)i * NCLASS + tid] = logits[((size_t)t * NB + b) * NCLASS + tid];
    float sc = invs[b * NT + t];
    const float* ap = A + ((size_t)b * NT + t) * NHW;
    float* op = out + (size_t)L * NCLASS + (size_t)i * NHW;
    for (int k = tid; k < NHW; k += blockDim.x) op[k] = ap[k] * sc;
}

// ---------------- host orchestration ----------------
extern "C" void kernel_launch(void* const* d_in, const int* in_sizes, int n_in,
                              void* d_out, int out_size, void* d_ws, size_t ws_size,
                              hipStream_t stream) {
    const float* feature = (const float*)d_in[0];
    const float* A       = (const float*)d_in[1];
    const int*   text    = (const int*)d_in[2];
    const int*   pack_b  = (const int*)d_in[4];
    const int*   pack_t  = (const int*)d_in[5];
    const float* emb     = (const float*)d_in[6];
    const float* Wih_f   = (const float*)d_in[7];
    const float* Whh_f   = (const float*)d_in[8];
    const float* bih_f   = (const float*)d_in[9];
    const float* bhh_f   = (const float*)d_in[10];
    const float* Wih_b   = (const float*)d_in[11];
    const float* Whh_b   = (const float*)d_in[12];
    const float* bih_b   = (const float*)d_in[13];
    const float* bhh_b   = (const float*)d_in[14];
    const float* Wih_g   = (const float*)d_in[15];
    const float* Whh_g   = (const float*)d_in[16];
    const float* bih_g   = (const float*)d_in[17];
    const float* bhh_g   = (const float*)d_in[18];
    const float* Wgen    = (const float*)d_in[19];
    const float* bgen    = (const float*)d_in[20];

    int L = in_sizes[4];  // lenText

    char* ws = (char*)d_ws;
    size_t off = 0;
    auto alloc = [&](size_t bytes) -> void* {
        void* p = ws + off;
        off = (off + bytes + 255) & ~(size_t)255;
        return p;
    };
    float*          invs   = (float*)alloc((size_t)NB * NT * 4);
    unsigned short* Cbf    = (unsigned short*)alloc((size_t)NT * NB * NC * 2);
    unsigned short* xs     = (unsigned short*)alloc((size_t)NT * NB * 2 * NC * 2);
    float*          Xproj  = (float*)alloc((size_t)2 * NT * NB * 512 * 4);
    float*          gxbuf  = (float*)alloc((size_t)NT * NB * 768 * 4);
    unsigned short* hsO    = (unsigned short*)alloc((size_t)NT * NB * NC * 2);
    float*          logits = (float*)alloc((size_t)NT * NB * NCLASS * 4);
    unsigned short* WihF   = (unsigned short*)alloc((size_t)512 * NC * 2);
    unsigned short* WihB   = (unsigned short*)alloc((size_t)512 * NC * 2);
    unsigned short* WhhL   = (unsigned short*)alloc((size_t)2 * 512 * NHID * 2);
    unsigned short* WihG   = (unsigned short*)alloc((size_t)768 * 512 * 2);
    unsigned short* WhhG   = (unsigned short*)alloc((size_t)768 * NC * 2);
    unsigned short* WgenB  = (unsigned short*)alloc((size_t)NCLASS * NC * 2);
    float*          biasL  = (float*)alloc((size_t)1024 * 4);
    (void)ws_size; (void)n_in; (void)out_size;

    auto cvt = [&](const float* s, unsigned short* d, int n) {
        k_convert<<<(n + 255) / 256, 256, 0, stream>>>(s, d, n);
    };
    cvt(Wih_f, WihF, 512 * NC);
    cvt(Wih_b, WihB, 512 * NC);
    cvt(Whh_f, WhhL, 512 * NHID);
    cvt(Whh_b, WhhL + 512 * NHID, 512 * NHID);
    cvt(Wih_g, WihG, 768 * 512);
    cvt(Whh_g, WhhG, 768 * NC);
    cvt(Wgen, WgenB, NCLASS * NC);
    k_lstm_bias<<<4, 256, 0, stream>>>(bih_f, bhh_f, bih_b, bhh_b, biasL);
    k_prev<<<(NT * NB * NC + 255) / 256, 256, 0, stream>>>(emb, text, xs);
    k_invsum<<<NB * NT / 8, 256, 0, stream>>>(A, invs);

    // attention pooling -> C bf16 [t][b][c]; one wave per 16x64 strip
    k_pool<<<NB * 16 / 8, 256, 0, stream>>>(feature, A, invs, Cbf);

    // LSTM input projections for all timesteps (M = NT*NB = 8192)
    k_gemm<<<(8192 / 16) * (512 / 64) / 8, 256, 0, stream>>>(Cbf, WihF, biasL, Xproj,
                                                             8192, 512, NC);
    k_gemm<<<(8192 / 16) * (512 / 64) / 8, 256, 0, stream>>>(Cbf, WihB, biasL + 512,
                                                             Xproj + (size_t)NT * NB * 512,
                                                             8192, 512, NC);
    // bidirectional LSTM recurrence -> lower half of xs
    k_lstm<<<2, 1024, 0, stream>>>(Xproj, WhhL, xs);

    // GRU input projection: gx = xs @ Wih_g^T + bih_g
    k_gemm<<<(8192 / 16) * (768 / 64) / 8, 256, 0, stream>>>(xs, WihG, bih_g, gxbuf,
                                                             8192, 768, 2 * NC);
    // GRU recurrence -> hs bf16
    k_gru<<<1, 1024, 0, stream>>>(gxbuf, WhhG, bhh_g, hsO);

    // generator: logits = hs @ Wgen^T + bgen
    k_gemm<<<(8192 / 16) * (NCLASS / 64) / 8, 256, 0, stream>>>(hsO, WgenB, bgen, logits,
                                                                8192, NCLASS, NC);
    // ragged gather of (out_res, out_attns)
    k_gather<<<L, 256, 0, stream>>>(logits, A, invs, pack_b, pack_t, (float*)d_out, L);
}
